// CgpHmmCell_78597901517185
// MI455X (gfx1250) — matrix operand order repaired and checked by reading, not compiled, not run on hardware
//
#include <hip/hip_runtime.h>
#include <hip/hip_bf16.h>
#include <math.h>

// ---------------- problem constants ----------------
#define N_STATES 612
#define SP       640                 // states padded: 20 K-tiles of 32
#define N_EMIS   126
#define BATCH    64
#define SEQ_LEN  3000
#define NWG      20                  // persistent workgroups, 32 A-columns each
#define COLS_PER_WG 32
#define KT_COUNT 20                  // 640 / 32
#define SCALEF   16384.0f            // 2^14 keeps unnormalized f16 alphas ~0.2
#define LOG_SCALE 9.7040605278392343f
#define TOTAL_LOG_SCALE (3000.0f * 9.7040605278392343f)

typedef __attribute__((ext_vector_type(16))) _Float16 v16h;
typedef __attribute__((ext_vector_type(8)))  _Float16 v8h;
typedef __attribute__((ext_vector_type(8)))  float    v8f;

// ---------------- workspace layout (bytes) ----------------
#define OFF_AT    0                        // A^T f16 [640 n][640 k]   = 819200 B
#define OFF_BEM   819200                   // Bemis f32 [126][640]     = 322560 B
#define OFF_I     1141760                  // I f32 [640]              = 2560 B
#define OFF_ALPHA 1144320                  // f16 [2][64][640]         = 163840 B
#define OFF_BAR   1308160                  // u32 [2] {count, epoch}
#define ZERO_BYTES 1308176                 // zero everything above
#define OFF_OBST  1308672                  // int [3000][64]           = 768000 B
#define WS_BYTES  2076672

// ---------------- zero scratch ----------------
__global__ void k_zero(unsigned* __restrict__ p, int n) {
  int i = blockIdx.x * blockDim.x + threadIdx.x;
  if (i < n) p[i] = 0u;
}

// ---------------- obsT[t][b] = obs[b][t] ----------------
__global__ __launch_bounds__(256) void k_tobs(const int* __restrict__ obs,
                                              int* __restrict__ obsT) {
  int i = blockIdx.x * blockDim.x + threadIdx.x;     // over BATCH*SEQ_LEN
  if (i < BATCH * SEQ_LEN) {
    int b = i / SEQ_LEN, t = i - b * SEQ_LEN;
    obsT[t * BATCH + b] = obs[i];
  }
}

// ---------------- A = softmax(rows of transition_kernel), stored transposed f16 ----------------
__global__ __launch_bounds__(256) void k_prep_A(const float* __restrict__ Tk,
                                                _Float16* __restrict__ At) {
  __shared__ float red[256];
  const int r = blockIdx.x;            // source state k = r
  const int tid = threadIdx.x;
  const float* row = Tk + r * N_STATES;

  float m = -3.4e38f;
  for (int j = tid; j < N_STATES; j += 256) m = fmaxf(m, row[j]);
  red[tid] = m; __syncthreads();
  for (int s = 128; s > 0; s >>= 1) { if (tid < s) red[tid] = fmaxf(red[tid], red[tid + s]); __syncthreads(); }
  const float rmax = red[0]; __syncthreads();

  float sum = 0.f;
  for (int j = tid; j < N_STATES; j += 256) sum += __expf(row[j] - rmax);
  red[tid] = sum; __syncthreads();
  for (int s = 128; s > 0; s >>= 1) { if (tid < s) red[tid] += red[tid + s]; __syncthreads(); }
  const float inv = 1.0f / red[0];

  for (int j = tid; j < N_STATES; j += 256)
    At[j * SP + r] = (_Float16)(__expf(row[j] - rmax) * inv);   // At[n][k] = A[k][n]
}

// ---------------- Bemis[e][s] = softmax over e of emission_kernel[e][s] ----------------
__global__ __launch_bounds__(128) void k_prep_B(const float* __restrict__ Ek,
                                                float* __restrict__ Bem) {
  __shared__ float red[128];
  const int s = blockIdx.x;            // state
  const int tid = threadIdx.x;
  const float v = (tid < N_EMIS) ? Ek[tid * N_STATES + s] : -3.4e38f;
  red[tid] = v; __syncthreads();
  for (int st = 64; st > 0; st >>= 1) { if (tid < st) red[tid] = fmaxf(red[tid], red[tid + st]); __syncthreads(); }
  const float m = red[0]; __syncthreads();
  const float e = (tid < N_EMIS) ? __expf(v - m) : 0.f;
  red[tid] = e; __syncthreads();
  for (int st = 64; st > 0; st >>= 1) { if (tid < st) red[tid] += red[tid + st]; __syncthreads(); }
  if (tid < N_EMIS) Bem[tid * SP + s] = e / red[0];
}

// ---------------- I = softmax(init_kernel), zero-padded ----------------
__global__ __launch_bounds__(256) void k_prep_I(const float* __restrict__ Ik,
                                                float* __restrict__ I) {
  __shared__ float red[256];
  const int tid = threadIdx.x;
  float m = -3.4e38f;
  for (int j = tid; j < N_STATES; j += 256) m = fmaxf(m, Ik[j]);
  red[tid] = m; __syncthreads();
  for (int s = 128; s > 0; s >>= 1) { if (tid < s) red[tid] = fmaxf(red[tid], red[tid + s]); __syncthreads(); }
  const float rmax = red[0]; __syncthreads();
  float sum = 0.f;
  for (int j = tid; j < N_STATES; j += 256) sum += __expf(Ik[j] - rmax);
  red[tid] = sum; __syncthreads();
  for (int s = 128; s > 0; s >>= 1) { if (tid < s) red[tid] += red[tid + s]; __syncthreads(); }
  const float inv = 1.0f / red[0];
  for (int j = tid; j < SP; j += 256)
    I[j] = (j < N_STATES) ? __expf(Ik[j] - rmax) * inv : 0.f;
}

// ---------------- alpha0 = I * Bemis[obs[:,0]] * SCALE (unnormalized) ----------------
__global__ __launch_bounds__(256) void k_alpha0(const int* __restrict__ obs,
                                                const float* __restrict__ I,
                                                const float* __restrict__ Bem,
                                                _Float16* __restrict__ alpha) {
  const int b = blockIdx.x, tid = threadIdx.x;
  const int o = obs[b * SEQ_LEN];
  for (int s = tid; s < SP; s += 256) {
    const float v = I[s] * Bem[o * SP + s] * SCALEF;   // padded s -> I[s]==0 -> v==0
    alpha[b * SP + s] = (_Float16)v;
  }
}

// ---------------- persistent forward-scan kernel: 20 WGs, A chunk LDS-resident ----------------
__global__ __launch_bounds__(256) void k_forward(const int* __restrict__ obsT,    // [3000][64]
                                                 const _Float16* __restrict__ At,
                                                 const float* __restrict__ Bem,
                                                 _Float16* __restrict__ alpha,     // [2][64][640]
                                                 unsigned* __restrict__ bar,       // {count, epoch}
                                                 float* __restrict__ out) {
  __shared__ __attribute__((aligned(32))) _Float16 Ach[COLS_PER_WG * SP];  // 40 KB, [col][k]

  const int tid = threadIdx.x;
  const int wg  = blockIdx.x;
  const int colbase = wg * COLS_PER_WG;

  // stage this WG's 32-column chunk of A (transposed layout -> contiguous copies)
  for (int i = tid; i < COLS_PER_WG * (SP / 8); i += 256) {
    const int c  = i / (SP / 8);
    const int kv = (i % (SP / 8)) * 8;
    *(v8h*)&Ach[c * SP + kv] = *(const v8h*)&At[(colbase + c) * SP + kv];
  }
  __syncthreads();

  const int lane = tid & 31;
  const int wv   = tid >> 5;       // 8 waves
  const int mt   = wv & 3;         // batch tile 0..3 (M = 64)
  const int nt   = wv >> 2;        // column tile 0..1 within chunk (N = 32)
  const int hi   = lane >> 4;
  const int l15  = lane & 15;

  const int arow  = mt * 16 + l15;                      // A-operand row (batch index)
  const int ncol  = nt * 16 + l15;                      // B-operand column within chunk
  const int nglob = colbase + ncol;
  const int mbase = mt * 16 + hi * 8;                   // C/D layout: M = mbase + r
  const _Float16* bbase = &Ach[ncol * SP + hi * 16];    // B: K = e + 16*hi, contiguous in k

  v16h onesv;
  #pragma unroll
  for (int i = 0; i < 16; ++i) onesv[i] = (_Float16)1.0f;

  float ll[8];
  #pragma unroll
  for (int r = 0; r < 8; ++r) ll[r] = 0.f;

  for (int t = 1; t < SEQ_LEN; ++t) {
    const int pb = (t - 1) & 1, nb = t & 1;

    // ---- early per-row emission gathers (pre-scaled): hidden under the WMMA K-loop ----
    float evs[8];
    #pragma unroll
    for (int r = 0; r < 8; ++r)
      evs[r] = Bem[obsT[t * BATCH + mbase + r] * SP + nglob] * SCALEF;

    if (t + 1 < SEQ_LEN)                                 // warm next step's emission rows
      __builtin_prefetch(&Bem[obsT[(t + 1) * BATCH + arow] * SP + colbase], 0, 1);

    // ---- P = V_{t-1} @ A_chunk  and  w_{t-1} = V_{t-1} @ ones  (f16 WMMA, f32 acc) ----
    const _Float16* abase = alpha + pb * BATCH * SP + arow * SP + hi * 16;  // A: K=e+16*hi
    v8f acc  = {0.f, 0.f, 0.f, 0.f, 0.f, 0.f, 0.f, 0.f};
    v8f accw = {0.f, 0.f, 0.f, 0.f, 0.f, 0.f, 0.f, 0.f};
    #pragma unroll 5
    for (int kt = 0; kt < KT_COUNT; ++kt) {
      const v8h alo = *(const v8h*)(abase + kt * 32);
      const v8h ahi = *(const v8h*)(abase + kt * 32 + 8);
      const v16h av = __builtin_shufflevector(alo, ahi, 0,1,2,3,4,5,6,7,8,9,10,11,12,13,14,15);
      const v16h bv = *(const v16h*)(bbase + kt * 32);
      acc  = __builtin_amdgcn_wmma_f32_16x16x32_f16(false, av, false, bv,    (short)0, acc,  false, false);
      accw = __builtin_amdgcn_wmma_f32_16x16x32_f16(false, av, false, onesv, (short)0, accw, false, false);
    }

    // ---- scale by emission * SCALE / w_{t-1} (delayed normalization, fast rcp), store ----
    _Float16* anext = alpha + nb * BATCH * SP + nglob;
    #pragma unroll
    for (int r = 0; r < 8; ++r) {
      const float v = acc[r] * evs[r] * __builtin_amdgcn_rcpf(accw[r]);
      anext[(mbase + r) * SP] = (_Float16)v;
    }
    if (wg == 0) {                                      // log z_{t-1} per row, in registers
      #pragma unroll
      for (int r = 0; r < 8; ++r) ll[r] += __logf(accw[r]);
    }

    // ---- grid-wide sense barrier (20 resident WGs) ----
    __threadfence();
    __syncthreads();
    if (tid == 0) {
      const unsigned e0 = __hip_atomic_load(&bar[1], __ATOMIC_RELAXED, __HIP_MEMORY_SCOPE_AGENT);
      const unsigned prev = __hip_atomic_fetch_add(&bar[0], 1u, __ATOMIC_ACQ_REL, __HIP_MEMORY_SCOPE_AGENT);
      if (prev == NWG - 1) {
        __hip_atomic_store(&bar[0], 0u, __ATOMIC_RELAXED, __HIP_MEMORY_SCOPE_AGENT);
        __hip_atomic_fetch_add(&bar[1], 1u, __ATOMIC_RELEASE, __HIP_MEMORY_SCOPE_AGENT);
      } else {
        while (__hip_atomic_load(&bar[1], __ATOMIC_ACQUIRE, __HIP_MEMORY_SCOPE_AGENT) == e0)
          __builtin_amdgcn_s_sleep(2);
      }
    }
    __syncthreads();
    __threadfence();
  }

  // ---- final: w_{T-1} via one more ones-WMMA pass; emit log-likelihood ----
  if (wg == 0) {
    const _Float16* abase = alpha + ((SEQ_LEN - 1) & 1) * BATCH * SP + arow * SP + hi * 8;
    v8f accw = {0.f, 0.f, 0.f, 0.f, 0.f, 0.f, 0.f, 0.f};
    #pragma unroll 5
    for (int kt = 0; kt < KT_COUNT; ++kt) {
      const v8h alo = *(const v8h*)(abase + kt * 32);
      const v8h ahi = *(const v8h*)(abase + kt * 32 + 16);
      const v16h av = __builtin_shufflevector(alo, ahi, 0,1,2,3,4,5,6,7,8,9,10,11,12,13,14,15);
      accw = __builtin_amdgcn_wmma_f32_16x16x32_f16(false, av, false, onesv, (short)0, accw, false, false);
    }
    if (nt == 0 && l15 == 0) {
      #pragma unroll
      for (int r = 0; r < 8; ++r)
        out[mbase + r] = ll[r] + __logf(accw[r]) - TOTAL_LOG_SCALE;
    }
  }
}

// ---------------- host launch ----------------
extern "C" void kernel_launch(void* const* d_in, const int* in_sizes, int n_in,
                              void* d_out, int out_size, void* d_ws, size_t ws_size,
                              hipStream_t stream) {
  const int*   obs = (const int*)d_in[0];
  const float* Tk  = (const float*)d_in[1];
  const float* Ek  = (const float*)d_in[2];
  const float* Ik  = (const float*)d_in[3];
  float* out = (float*)d_out;

  char* ws = (char*)d_ws;
  _Float16* At    = (_Float16*)(ws + OFF_AT);
  float*    Bem   = (float*)   (ws + OFF_BEM);
  float*    I     = (float*)   (ws + OFF_I);
  _Float16* alpha = (_Float16*)(ws + OFF_ALPHA);
  unsigned* bar   = (unsigned*)(ws + OFF_BAR);
  int*      obsT  = (int*)     (ws + OFF_OBST);

  const int zwords = ZERO_BYTES / 4;
  k_zero  <<<(zwords + 255) / 256, 256, 0, stream>>>((unsigned*)ws, zwords);
  k_tobs  <<<(BATCH * SEQ_LEN + 255) / 256, 256, 0, stream>>>(obs, obsT);
  k_prep_A<<<N_STATES, 256, 0, stream>>>(Tk, At);
  k_prep_B<<<N_STATES, 128, 0, stream>>>(Ek, Bem);
  k_prep_I<<<1, 256, 0, stream>>>(Ik, I);
  k_alpha0<<<BATCH, 256, 0, stream>>>(obs, I, Bem, alpha);
  k_forward<<<NWG, 256, 0, stream>>>(obsT, At, Bem, alpha, bar, out);
}
